// Decoder_8581344657802
// MI455X (gfx1250) — matrix operand order repaired and checked by reading, not compiled, not run on hardware
//
#include <hip/hip_runtime.h>
#include <hip/hip_bf16.h>
#include <math.h>

// ---------------------------------------------------------------------------
// Types for CDNA5 WMMA (gfx1250, wave32)
// ---------------------------------------------------------------------------
typedef __attribute__((ext_vector_type(16))) __bf16 v16bf;
typedef __attribute__((ext_vector_type(8)))  float  v8f;
typedef __attribute__((ext_vector_type(8)))  __bf16 bf16x8;

union V16U { v16bf v; bf16x8 h[2]; };

// Problem constants (match reference)
#define BB   32
#define SS   64
#define VV   32000
#define EMB_ 512
#define EE   512          // encoder hidden (2E = 1024)
#define DD   1024
#define TWOE 1024
#define XDIM 1536         // EMB + 2E
#define G3D  3072         // 3*D

// ---------------------------------------------------------------------------
// Generic  C[M,N] = A[M,K] @ W[N,K]^T (+bias)   — bf16 in, f32 out, WMMA
// Each wave: 32x16 output tile (two 16x16 f32 accumulators share B fragment).
// grid.x * 8 waves tile N (16 cols/wave), grid.y tiles M (32 rows/block).
// K must be a multiple of 32; M multiple of 32; N multiple of 128.
// ---------------------------------------------------------------------------
__global__ __launch_bounds__(256) void gemm_xwt_bf16_wmma(
    const __bf16* __restrict__ A, int lda,
    const __bf16* __restrict__ W, int ldw,
    float* __restrict__ C, long long ldc,
    const float* __restrict__ bias, int K)
{
    const int wave = threadIdx.x >> 5;
    const int lane = threadIdx.x & 31;
    const int n0   = (blockIdx.x * 8 + wave) * 16;
    const int m0   = blockIdx.y * 32;

    // A fragment layout (16-bit A 16x32, ISA 7.12.2):
    //  lane L holds row M = L&15; K groups: lanes 0-15 -> K 0-7 & 16-23,
    //  lanes 16-31 -> K 8-15 & 24-31.
    const int rowA = lane & 15;
    const int kgA  = (lane >> 4) * 8;
    // B fragment layout (16-bit B 32x16): lane L holds col N = L&15;
    //  lanes 0-15 -> K 0-7 & 16-23, lanes 16-31 -> K 8-15 & 24-31 (same as A).
    const int col  = lane & 15;
    const int kgB  = (lane >> 4) * 8;

    const __bf16* a0p = A + (size_t)(m0 + rowA)      * lda;
    const __bf16* a1p = A + (size_t)(m0 + 16 + rowA) * lda;
    const __bf16* bp  = W + (size_t)(n0 + col)       * ldw;

    v8f acc0 = {};
    v8f acc1 = {};

    for (int k0 = 0; k0 < K; k0 += 32) {
        V16U a0, a1, bb;
        a0.h[0] = *(const bf16x8*)(a0p + k0 + kgA);
        a0.h[1] = *(const bf16x8*)(a0p + k0 + 16 + kgA);
        a1.h[0] = *(const bf16x8*)(a1p + k0 + kgA);
        a1.h[1] = *(const bf16x8*)(a1p + k0 + 16 + kgA);
        bb.h[0] = *(const bf16x8*)(bp  + k0 + kgB);
        bb.h[1] = *(const bf16x8*)(bp  + k0 + 16 + kgB);
        acc0 = __builtin_amdgcn_wmma_f32_16x16x32_bf16(
                   false, a0.v, false, bb.v, (short)0, acc0, false, false);
        acc1 = __builtin_amdgcn_wmma_f32_16x16x32_bf16(
                   false, a1.v, false, bb.v, (short)0, acc1, false, false);
    }

    // C/D layout: VGPR r -> M = r (lanes 0-15) or 8+r (lanes 16-31); N = lane&15
    const int n     = n0 + col;
    const float bv  = bias ? bias[n] : 0.0f;
    const int mBase = (lane >> 4) * 8;
#pragma unroll
    for (int r = 0; r < 8; ++r) {
        const int m = m0 + mBase + r;
        C[(size_t)m        * ldc + n] = acc0[r] + bv;
        C[(size_t)(m + 16) * ldc + n] = acc1[r] + bv;
    }
}

// ---------------------------------------------------------------------------
// One-time data movement / precision conversion kernels
// ---------------------------------------------------------------------------
__global__ void f32_to_bf16_kernel(const float* __restrict__ in,
                                   __bf16* __restrict__ out, size_t n)
{
    size_t i      = blockIdx.x * (size_t)blockDim.x + threadIdx.x;
    size_t stride = (size_t)gridDim.x * blockDim.x;
    for (; i < n; i += stride) out[i] = (__bf16)in[i];
}

// out[c*rows + r] = in[r*cols + c]  (produces W^T in [N,K] "weight" layout)
__global__ void transpose_f32_to_bf16_kernel(const float* __restrict__ in,
                                             __bf16* __restrict__ out,
                                             int rows, int cols)
{
    size_t i      = blockIdx.x * (size_t)blockDim.x + threadIdx.x;
    size_t total  = (size_t)rows * cols;
    size_t stride = (size_t)gridDim.x * blockDim.x;
    for (; i < total; i += stride) {
        int r = (int)(i / cols), c = (int)(i % cols);
        out[(size_t)c * rows + r] = (__bf16)in[i];
    }
}

__global__ void init_h_kernel(const float* __restrict__ h0,
                              float* __restrict__ h,
                              __bf16* __restrict__ hb, int n)
{
    int i = blockIdx.x * blockDim.x + threadIdx.x;
    if (i < n) { float v = h0[i]; h[i] = v; hb[i] = (__bf16)v; }
}

// d_out[:, 0, :] = one-hot(index 0)
__global__ void init_out0_kernel(float* __restrict__ out)
{
    size_t i     = blockIdx.x * (size_t)blockDim.x + threadIdx.x;
    size_t total = (size_t)BB * VV;
    if (i < total) {
        int b = (int)(i / VV), j = (int)(i % VV);
        out[(size_t)b * SS * VV + j] = (j == 0) ? 1.0f : 0.0f;
    }
}

// ---------------------------------------------------------------------------
// Attention: scores = v_a . tanh(enc_proj + hUa), masked softmax over S,
// ctx = sum_s w_s * encoder_outs[b,s,:].  One block per batch row.
// Writes ctx (bf16) into x_bf16[b, EMB: EMB+2E].
// ---------------------------------------------------------------------------
__global__ __launch_bounds__(256) void attn_ctx_kernel(
    const float* __restrict__ enc_proj,   // [B,S,D]
    const float* __restrict__ hUa,        // [B,D]
    const float* __restrict__ va,         // [D]
    const float* __restrict__ enc_outs,   // [B,S,2E]
    const int*   __restrict__ lengths,    // [B]
    __bf16* __restrict__ x_bf16)          // [B, XDIM]
{
    const int b    = blockIdx.x;
    const int tid  = threadIdx.x;
    const int lane = tid & 31;
    const int wave = tid >> 5;
    __shared__ float scores[SS];

    const int len = lengths[b];
    const float* hrow = hUa + (size_t)b * DD;

    for (int i = 0; i < 8; ++i) {               // 8 waves x 8 = 64 positions
        const int s = wave * 8 + i;
        const float* ep = enc_proj + ((size_t)b * SS + s) * DD;
        float acc = 0.0f;
        for (int d = lane; d < DD; d += 32)
            acc += tanhf(ep[d] + hrow[d]) * va[d];
        for (int off = 16; off > 0; off >>= 1)
            acc += __shfl_down(acc, off, 32);
        if (lane == 0) scores[s] = (s < len) ? acc : -INFINITY;
    }
    __syncthreads();

    // redundant per-thread softmax stats over 64 values (cheap)
    float mx = -INFINITY;
    for (int s = 0; s < SS; ++s) mx = fmaxf(mx, scores[s]);
    float sum = 0.0f;
    for (int s = 0; s < SS; ++s) sum += expf(scores[s] - mx);
    const float inv = 1.0f / sum;

    for (int e = tid; e < TWOE; e += 256) {
        float acc = 0.0f;
        for (int s = 0; s < SS; ++s) {
            const float w = expf(scores[s] - mx) * inv;
            acc += w * enc_outs[((size_t)b * SS + s) * TWOE + e];
        }
        x_bf16[(size_t)b * XDIM + EMB_ + e] = (__bf16)acc;
    }
}

// ---------------------------------------------------------------------------
// Input select: greedy = argmax(prev logits at position t), teacher = tokens.
// coin from a deterministic hash (JAX threefry not replicable on-device;
// with teacher_forcing_ratio = 1.0 the selected branch is identical).
// Writes chosen embedding row (bf16) into x_bf16[b, 0:EMB].
// ---------------------------------------------------------------------------
__global__ __launch_bounds__(256) void select_input_kernel(
    const float* __restrict__ d_out_all,  // [B,S,V]
    int t,
    const int*   __restrict__ tokens,     // [B,S-1]
    const float* __restrict__ embedding,  // [V,EMB]
    const float* __restrict__ tfr,        // scalar
    __bf16* __restrict__ x_bf16)          // [B,XDIM]
{
    const int b = blockIdx.x;
    const float* prev = d_out_all + ((size_t)b * SS + t) * VV;

    __shared__ float sv[256];
    __shared__ int   si[256];
    float bv = -INFINITY; int bi = 0;
    for (int j = threadIdx.x; j < VV; j += 256) {
        const float v = prev[j];
        if (v > bv) { bv = v; bi = j; }
    }
    sv[threadIdx.x] = bv; si[threadIdx.x] = bi;
    __syncthreads();
    for (int off = 128; off > 0; off >>= 1) {
        if (threadIdx.x < off) {
            const float v2 = sv[threadIdx.x + off];
            const int   i2 = si[threadIdx.x + off];
            if (v2 > sv[threadIdx.x] ||
                (v2 == sv[threadIdx.x] && i2 < si[threadIdx.x])) {
                sv[threadIdx.x] = v2; si[threadIdx.x] = i2;
            }
        }
        __syncthreads();
    }
    const int amax = si[0];

    unsigned x = (unsigned)t * 0x9E3779B9u + 0x85EBCA6Bu;
    x ^= x >> 16; x *= 0x7FEB352Du; x ^= x >> 15; x *= 0x846CA68Bu; x ^= x >> 16;
    const float coin = (float)(x >> 8) * (1.0f / 16777216.0f);

    const int tok = (coin < tfr[0]) ? tokens[b * (SS - 1) + t] : amax;
    const float* erow = embedding + (size_t)tok * EMB_;
    for (int e = threadIdx.x; e < EMB_; e += 256)
        x_bf16[(size_t)b * XDIM + e] = (__bf16)erow[e];
}

// ---------------------------------------------------------------------------
// GRU gate fusion: h = (1-z)*n + z*h
// ---------------------------------------------------------------------------
__global__ __launch_bounds__(256) void gru_update_kernel(
    const float* __restrict__ gx,   // [B,3D]
    const float* __restrict__ gh,   // [B,3D]
    const float* __restrict__ b_ih, // [3D]
    const float* __restrict__ b_hh, // [3D]
    float*  __restrict__ h,         // [B,D] in/out
    __bf16* __restrict__ h_bf16)    // [B,D]
{
    const int idx = blockIdx.x * blockDim.x + threadIdx.x;  // over B*D
    const int b = idx >> 10, d = idx & (DD - 1);
    const float* gxr = gx + (size_t)b * G3D;
    const float* ghr = gh + (size_t)b * G3D;

    float r = gxr[d]        + b_ih[d]        + ghr[d]        + b_hh[d];
    r = 1.0f / (1.0f + expf(-r));
    float z = gxr[DD + d]   + b_ih[DD + d]   + ghr[DD + d]   + b_hh[DD + d];
    z = 1.0f / (1.0f + expf(-z));
    const float n = tanhf(gxr[2*DD + d] + b_ih[2*DD + d] +
                          r * (ghr[2*DD + d] + b_hh[2*DD + d]));
    const float hn = (1.0f - z) * n + z * h[idx];
    h[idx]      = hn;
    h_bf16[idx] = (__bf16)hn;
}

// ---------------------------------------------------------------------------
// Host launcher
// ---------------------------------------------------------------------------
extern "C" void kernel_launch(void* const* d_in, const int* in_sizes, int n_in,
                              void* d_out, int out_size, void* d_ws, size_t ws_size,
                              hipStream_t stream)
{
    const int*   tokens    = (const int*)  d_in[0];
    const float* hidden0   = (const float*)d_in[1];
    const float* enc_outs  = (const float*)d_in[2];
    const int*   lengths   = (const int*)  d_in[3];
    const float* tfr       = (const float*)d_in[4];
    const float* embedding = (const float*)d_in[5];
    const float* Wa        = (const float*)d_in[6];
    const float* Ua        = (const float*)d_in[7];
    const float* v_a       = (const float*)d_in[8];
    const float* W_ih      = (const float*)d_in[9];
    const float* b_ih      = (const float*)d_in[10];
    const float* W_hh      = (const float*)d_in[11];
    const float* b_hh      = (const float*)d_in[12];
    const float* W_out     = (const float*)d_in[13];
    const float* b_out     = (const float*)d_in[14];
    float* out = (float*)d_out;

    char*  ws  = (char*)d_ws;
    size_t off = 0;
    auto alloc = [&](size_t bytes) -> void* {
        off = (off + 255) & ~(size_t)255;
        void* p = ws + off;
        off += bytes;
        return p;
    };

    __bf16* wout_b   = (__bf16*)alloc((size_t)VV  * DD   * 2);   // 65.5 MB
    __bf16* wih_b    = (__bf16*)alloc((size_t)G3D * XDIM * 2);   //  9.4 MB
    __bf16* whh_b    = (__bf16*)alloc((size_t)G3D * DD   * 2);   //  6.3 MB
    __bf16* uat_b    = (__bf16*)alloc((size_t)DD  * DD   * 2);   //  2.1 MB
    __bf16* wat_b    = (__bf16*)alloc((size_t)TWOE* DD   * 2);   //  2.1 MB
    __bf16* enc_b    = (__bf16*)alloc((size_t)BB * SS * TWOE * 2); // 4.2 MB
    float*  enc_proj = (float*) alloc((size_t)BB * SS * DD * 4);   // 8.4 MB
    float*  h_f32    = (float*) alloc((size_t)BB * DD * 4);
    __bf16* h_b      = (__bf16*)alloc((size_t)BB * DD * 2);
    float*  hUa      = (float*) alloc((size_t)BB * DD * 4);
    __bf16* x_b      = (__bf16*)alloc((size_t)BB * XDIM * 2);
    float*  gx       = (float*) alloc((size_t)BB * G3D * 4);
    float*  gh       = (float*) alloc((size_t)BB * G3D * 4);
    (void)ws_size; (void)in_sizes; (void)n_in; (void)out_size;

    // --- one-time precision conversion / transposes ---
    f32_to_bf16_kernel<<<2048, 256, 0, stream>>>(W_out, wout_b, (size_t)VV * DD);
    f32_to_bf16_kernel<<<512,  256, 0, stream>>>(W_ih,  wih_b,  (size_t)G3D * XDIM);
    f32_to_bf16_kernel<<<512,  256, 0, stream>>>(W_hh,  whh_b,  (size_t)G3D * DD);
    f32_to_bf16_kernel<<<256,  256, 0, stream>>>(enc_outs, enc_b, (size_t)BB * SS * TWOE);
    transpose_f32_to_bf16_kernel<<<4096, 256, 0, stream>>>(Ua, uat_b, DD, DD);
    transpose_f32_to_bf16_kernel<<<4096, 256, 0, stream>>>(Wa, wat_b, TWOE, DD);
    init_h_kernel<<<(BB * DD) / 256, 256, 0, stream>>>(hidden0, h_f32, h_b, BB * DD);
    init_out0_kernel<<<((size_t)BB * VV + 255) / 256, 256, 0, stream>>>(out);

    // enc_proj[B*S, D] = enc_bf16[B*S, 2E] @ Wa^T[ D, 2E ]^T   (i.e. enc @ Wa)
    gemm_xwt_bf16_wmma<<<dim3(DD / 128, (BB * SS) / 32), 256, 0, stream>>>(
        enc_b, TWOE, wat_b, TWOE, enc_proj, DD, nullptr, TWOE);

    // --- 63 sequential decode steps ---
    for (int t = 0; t < SS - 1; ++t) {
        // hUa = h @ Ua   (Ua pre-transposed into [N,K] weight layout)
        gemm_xwt_bf16_wmma<<<dim3(DD / 128, 1), 256, 0, stream>>>(
            h_b, DD, uat_b, DD, hUa, DD, nullptr, DD);

        attn_ctx_kernel<<<BB, 256, 0, stream>>>(
            enc_proj, hUa, v_a, enc_outs, lengths, x_b);

        select_input_kernel<<<BB, 256, 0, stream>>>(
            out, t, tokens, embedding, tfr, x_b);

        // gx = x @ W_ih^T ; gh = h @ W_hh^T   (biases fused in GRU kernel)
        gemm_xwt_bf16_wmma<<<dim3(G3D / 128, 1), 256, 0, stream>>>(
            x_b, XDIM, wih_b, XDIM, gx, G3D, nullptr, XDIM);
        gemm_xwt_bf16_wmma<<<dim3(G3D / 128, 1), 256, 0, stream>>>(
            h_b, DD, whh_b, DD, gh, G3D, nullptr, DD);

        gru_update_kernel<<<(BB * DD) / 256, 256, 0, stream>>>(
            gx, gh, b_ih, b_hh, h_f32, h_b);

        // logits -> d_out[:, t+1, :]  (bias fused; ldc = S*V batch stride)
        gemm_xwt_bf16_wmma<<<dim3(VV / 128, 1), 256, 0, stream>>>(
            h_b, DD, wout_b, DD, out + (size_t)(t + 1) * VV,
            (long long)SS * VV, b_out, DD);
    }
}
